// SimpleTransformer_83803401880032
// MI455X (gfx1250) — compile-verified, matrix operand order checked by reading, not run on hardware
//
#include <hip/hip_runtime.h>

// ---------------- model dims ----------------
#define LAYERS  4
#define DMODEL  1024
#define NHEADS  16
#define HDIM    64
#define FFNDIM  4096
#define BATCH   4
#define SEQ     1024
#define BAND    8
#define MROWS   (BATCH*SEQ)   // 4096
#define LN_EPS  1e-5f

// ---------------- vector types ----------------
typedef __attribute__((ext_vector_type(16))) __bf16 v16bf;
typedef __attribute__((ext_vector_type(8)))  __bf16 v8bf;
typedef __attribute__((ext_vector_type(8)))  float  v8f;
typedef __attribute__((ext_vector_type(4)))  float  v4f;
typedef __attribute__((ext_vector_type(4)))  unsigned int v4u;
typedef __attribute__((ext_vector_type(8)))  int v8i;
typedef __attribute__((ext_vector_type(4)))  int v4i;

// ---------------- GEMM tiling ----------------
// Block tile: 128 (M) x 256 (N) x 32 (K). 8 waves; each wave -> 64x64 via 4x4 WMMAs.
#define BM 128
#define BN 256
#define BK 32
#define LDT 40   // padded LDS row stride (elements) = 80B -> 20 banks, conflict-free

__device__ __forceinline__ float gelu_exact(float x) {
    return 0.5f * x * (1.0f + erff(x * 0.70710678118654752440f));
}

__device__ __forceinline__ v16bf cat8(v8bf lo, v8bf hi) {
    return __builtin_shufflevector(lo, hi, 0,1,2,3,4,5,6,7,8,9,10,11,12,13,14,15);
}

// ISA 7.12.2 bf16 fragment: lane<16 -> K0..7 & K16..23 ; lane>=16 -> K8..15 & K24..31
__device__ __forceinline__ v16bf frag_load(const __bf16* p) {
    v8bf lo = *(const v8bf*)p;          // +0B
    v8bf hi = *(const v8bf*)(p + 16);   // +32B
    return cat8(lo, hi);
}

// Async copy 16 bytes global -> LDS (per-lane, ASYNCcnt-tracked).
__device__ __forceinline__ void async_copy16(unsigned int lds_off, const __bf16* g) {
    unsigned long long ga = (unsigned long long)(uintptr_t)g;
    asm volatile("global_load_async_to_lds_b128 %0, %1, off"
                 :: "v"(lds_off), "v"(ga) : "memory");
}

// Tensor Data Mover: DMA a 2D tile [tile_rows x 32 bf16] of a [rows x Kd] bf16
// tensor into LDS at lds_addr, padding 16B after every 64B row -> LDT stride.
// D# layout per CDNA5 ISA ch.8 (group0 128b, group1 256b, groups 2/3 zero).
__device__ __forceinline__ void tdm_load_2d(unsigned int lds_addr, const __bf16* gptr,
                                            int rows, int Kd, int tile_rows)
{
    unsigned long long ga = (unsigned long long)(uintptr_t)gptr;
    v4u g0 = {0u, 0u, 0u, 0u};
    g0[0] = 1u;                                                    // count=1, no gather
    g0[1] = lds_addr;                                              // lds_addr [63:32]
    g0[2] = (unsigned int)ga;                                      // global_addr[31:0]
    g0[3] = (unsigned int)((ga >> 32) & 0x01FFFFFFull) | 0x80000000u; // addr[56:32] | type=2

    v8i g1 = {0,0,0,0,0,0,0,0};
    // data_size=1 (2B) | pad_enable | pad_interval=3 (16 DW) | pad_amount=3 (4 DW)
    g1[0] = (int)((1u << 16) | (1u << 20) | (3u << 22) | (3u << 25));
    g1[1] = (int)((unsigned)(Kd & 0xFFFF) << 16);                  // tensor_dim0[15:0]
    g1[2] = (int)(((unsigned)Kd >> 16) | ((unsigned)(rows & 0xFFFF) << 16)); // dim0 hi | dim1 lo
    g1[3] = (int)(((unsigned)rows >> 16) | (32u << 16));           // dim1 hi | tile_dim0=32
    g1[4] = tile_rows;                                             // tile_dim1 (tile_dim2=0)
    g1[5] = Kd;                                                    // tensor_dim0_stride[31:0]
    g1[6] = 0;
    g1[7] = 0;

    v4i g2 = {0,0,0,0};
    v4i g3 = {0,0,0,0};
#if defined(__clang_major__) && (__clang_major__ >= 23)
    v8i g4 = {0,0,0,0,0,0,0,0};
    __builtin_amdgcn_tensor_load_to_lds(g0, g1, g2, g3, g4, 0);
#else
    __builtin_amdgcn_tensor_load_to_lds(g0, g1, g2, g3, 0);
#endif
}

// C[M,N] = A[M,K] x W[N,K]^T + bias (optionally exact GELU). A, W in bf16.
// A tile: per-lane async-to-LDS. B tile: one TDM descriptor per tile (wave 0).
template<typename TO, bool GELU>
__global__ __launch_bounds__(256)
void gemm_wmma(const __bf16* __restrict__ A, const __bf16* __restrict__ W,
               const float* __restrict__ bias, TO* __restrict__ C,
               int M, int N, int Kd)
{
    __shared__ __align__(16) __bf16 sA[2][BM * LDT];   // 2 x 10 KB
    __shared__ __align__(16) __bf16 sB[2][BN * LDT];   // 2 x 20 KB

    const int t     = threadIdx.x;
    const int n0    = blockIdx.x * BN;
    const int m0    = blockIdx.y * BM;
    const int wave  = t >> 5;
    const int lane  = t & 31;
    const int waveM = wave >> 2;     // 0..1 -> 64-row slab
    const int waveN = wave & 3;      // 0..3 -> 64-col slab
    const int sel   = lane >> 4;     // half-wave selector
    const int rl    = lane & 15;

    // A staging mapping (per thread, per tile): one half-row = 2 x 16B
    const int arow  = t >> 1;        // 0..127
    const int ahalf = t & 1;         // 0/1
    const unsigned int laA0 = (unsigned int)(uintptr_t)&sA[0][arow * LDT + ahalf * 16];
    const unsigned int bufA = (unsigned int)(BM * LDT * sizeof(__bf16));
    const unsigned int ldsB[2] = { (unsigned int)(uintptr_t)&sB[0][0],
                                   (unsigned int)(uintptr_t)&sB[1][0] };
    const __bf16* gA = A + (size_t)(m0 + arow) * Kd + ahalf * 16;
    const __bf16* gB = W + (size_t)n0 * Kd;            // tile kt starts at gB + kt*BK

    v8f acc[4][4];
#pragma unroll
    for (int mi = 0; mi < 4; ++mi)
#pragma unroll
        for (int ni = 0; ni < 4; ++ni)
#pragma unroll
            for (int e = 0; e < 8; ++e) acc[mi][ni][e] = 0.0f;

    const int nk = Kd / BK;

    // prologue: stage tile 0 into buffer 0
    async_copy16(laA0,      gA);
    async_copy16(laA0 + 16, gA + 8);
    if (wave == 0) tdm_load_2d(ldsB[0], gB, N, Kd, BN);

    for (int kt = 0; kt < nk; ++kt) {
        const int buf = kt & 1;
        if (kt + 1 < nk) {
            // issue DMA for tile kt+1 into the other buffer
            const unsigned int la = laA0 + (buf ? 0u : bufA);
            const __bf16* ga = gA + (size_t)(kt + 1) * BK;
            async_copy16(la,      ga);
            async_copy16(la + 16, ga + 8);
            if (wave == 0) tdm_load_2d(ldsB[buf ^ 1], gB + (size_t)(kt + 1) * BK, N, Kd, BN);
            asm volatile("s_wait_asynccnt 0x2" ::: "memory");  // A tile kt complete
            __builtin_amdgcn_s_wait_tensorcnt(1);              // B tile kt complete (wave 0)
        } else {
            asm volatile("s_wait_asynccnt 0x0" ::: "memory");
            __builtin_amdgcn_s_wait_tensorcnt(0);
        }
        __syncthreads();

        v16bf afrag[4], bfrag[4];
#pragma unroll
        for (int mi = 0; mi < 4; ++mi)
            afrag[mi] = frag_load(&sA[buf][(waveM * 64 + mi * 16 + rl) * LDT + sel * 8]);
#pragma unroll
        for (int ni = 0; ni < 4; ++ni)
            bfrag[ni] = frag_load(&sB[buf][(waveN * 64 + ni * 16 + rl) * LDT + sel * 8]);

#pragma unroll
        for (int mi = 0; mi < 4; ++mi)
#pragma unroll
            for (int ni = 0; ni < 4; ++ni)
                acc[mi][ni] = __builtin_amdgcn_wmma_f32_16x16x32_bf16(
                    false, afrag[mi], false, bfrag[ni],
                    (short)0, acc[mi][ni], false, false);
        __syncthreads();   // everyone done reading buf before it is re-staged
    }

    // epilogue: C/D layout -> element (m = mbase + i, n), VGPR i, half-wave sel
#pragma unroll
    for (int ni = 0; ni < 4; ++ni) {
        const int n  = n0 + waveN * 64 + ni * 16 + rl;
        const float bn = bias[n];
#pragma unroll
        for (int mi = 0; mi < 4; ++mi) {
            const int mbase = m0 + waveM * 64 + mi * 16 + sel * 8;
#pragma unroll
            for (int i = 0; i < 8; ++i) {
                float v = acc[mi][ni][i] + bn;
                if (GELU) v = gelu_exact(v);
                C[(size_t)(mbase + i) * N + n] = (TO)v;
            }
        }
    }
}

// ---------------- f32 -> bf16 conversion (grid-stride, x4) ----------------
__global__ __launch_bounds__(256)
void cvt_f32_bf16(const float* __restrict__ in, __bf16* __restrict__ out, long n)
{
    long i = ((long)blockIdx.x * blockDim.x + threadIdx.x) * 4;
    const long stride = (long)gridDim.x * blockDim.x * 4;
    for (; i < n; i += stride) {
        v4f a = *(const v4f*)(in + i);
        out[i + 0] = (__bf16)a[0];
        out[i + 1] = (__bf16)a[1];
        out[i + 2] = (__bf16)a[2];
        out[i + 3] = (__bf16)a[3];
    }
}

// ---------------- banded attention: one WAVE per (b, h, q) ----------------
// lane owns head-dims (2*lane, 2*lane+1): coalesced 256B row reads,
// scores reduced with a 5-step shfl_xor butterfly (all lanes end with the sum).
__global__ __launch_bounds__(256)
void attn_band_kernel(const float* __restrict__ qkv, __bf16* __restrict__ o)
{
    const int wid  = (blockIdx.x * blockDim.x + threadIdx.x) >> 5;
    const int lane = threadIdx.x & 31;
    if (wid >= BATCH * NHEADS * SEQ) return;
    const int s = wid & (SEQ - 1);
    const int h = (wid >> 10) & (NHEADS - 1);
    const int b = wid >> 14;

    const size_t base = (size_t)(b * SEQ + s) * (3 * DMODEL);
    const int d0 = lane * 2;
    const float2 qv = *(const float2*)(qkv + base + h * HDIM + d0);

    int nk = SEQ - s; if (nk > BAND) nk = BAND;
    float sc[BAND];
    float mx = -3.4e38f;
    for (int j = 0; j < nk; ++j) {
        const float2 kv = *(const float2*)(qkv + base + (size_t)j * 3 * DMODEL
                                           + DMODEL + h * HDIM + d0);
        float p = qv.x * kv.x + qv.y * kv.y;
#pragma unroll
        for (int m = 16; m > 0; m >>= 1) p += __shfl_xor(p, m, 32);
        sc[j] = p * 0.125f;                    // 1/sqrt(64)
        mx = fmaxf(mx, sc[j]);
    }
    float denom = 0.0f;
    for (int j = 0; j < nk; ++j) { sc[j] = __expf(sc[j] - mx); denom += sc[j]; }
    const float inv = 1.0f / denom;

    float a0 = 0.0f, a1 = 0.0f;
    for (int j = 0; j < nk; ++j) {
        const float2 vv = *(const float2*)(qkv + base + (size_t)j * 3 * DMODEL
                                           + 2 * DMODEL + h * HDIM + d0);
        a0 += sc[j] * vv.x;
        a1 += sc[j] * vv.y;
    }
    __bf16* orow = o + (size_t)(b * SEQ + s) * DMODEL + h * HDIM + d0;
    orow[0] = (__bf16)(a0 * inv);
    orow[1] = (__bf16)(a1 * inv);
}

// ---------------- residual + LayerNorm, dual f32 + bf16 output ----------------
__global__ __launch_bounds__(256)
void ln_residual_kernel(const float* __restrict__ res, const float* __restrict__ y,
                        const float* __restrict__ g, const float* __restrict__ bta,
                        float* __restrict__ out, __bf16* __restrict__ outb)
{
    __shared__ float red[256];
    const int row = blockIdx.x;
    const int t   = threadIdx.x;
    const float* xr = res + (size_t)row * DMODEL;
    const float* yr = y   + (size_t)row * DMODEL;

    float v[4];
    float lsum = 0.0f;
#pragma unroll
    for (int i = 0; i < 4; ++i) { v[i] = xr[t + 256 * i] + yr[t + 256 * i]; lsum += v[i]; }
    red[t] = lsum; __syncthreads();
    for (int st = 128; st > 0; st >>= 1) { if (t < st) red[t] += red[t + st]; __syncthreads(); }
    const float mean = red[0] * (1.0f / DMODEL);
    __syncthreads();

    float lvar = 0.0f;
#pragma unroll
    for (int i = 0; i < 4; ++i) { float d = v[i] - mean; lvar += d * d; }
    red[t] = lvar; __syncthreads();
    for (int st = 128; st > 0; st >>= 1) { if (t < st) red[t] += red[t + st]; __syncthreads(); }
    const float rstd = rsqrtf(red[0] * (1.0f / DMODEL) + LN_EPS);

#pragma unroll
    for (int i = 0; i < 4; ++i) {
        const int c = t + 256 * i;
        const float ov = (v[i] - mean) * rstd * g[c] + bta[c];
        out [(size_t)row * DMODEL + c] = ov;
        outb[(size_t)row * DMODEL + c] = (__bf16)ov;
    }
}

// ---------------- orchestration ----------------
extern "C" void kernel_launch(void* const* d_in, const int* in_sizes, int n_in,
                              void* d_out, int out_size, void* d_ws, size_t ws_size,
                              hipStream_t stream)
{
    const float* src  = (const float*)d_in[0];
    const float* Wqkv = (const float*)d_in[1];
    const float* bqkv = (const float*)d_in[2];
    const float* Wo   = (const float*)d_in[3];
    const float* bo   = (const float*)d_in[4];
    const float* W1   = (const float*)d_in[5];
    const float* b1   = (const float*)d_in[6];
    const float* W2   = (const float*)d_in[7];
    const float* b2   = (const float*)d_in[8];
    const float* ln1w = (const float*)d_in[9];
    const float* ln1b = (const float*)d_in[10];
    const float* ln2w = (const float*)d_in[11];
    const float* ln2b = (const float*)d_in[12];

    float* x  = (float*)d_out;                       // running residual stream (f32)
    char*  ws = (char*)d_ws;
    const size_t MB = 1024 * 1024;
    __bf16* wbqkv = (__bf16*)(ws);                   // 24 MiB
    __bf16* wbo   = (__bf16*)(ws +  24 * MB);        //  8 MiB
    __bf16* wb1   = (__bf16*)(ws +  32 * MB);        // 32 MiB
    __bf16* wb2   = (__bf16*)(ws +  64 * MB);        // 32 MiB
    float*  qkv   = (float*) (ws +  96 * MB);        // 48 MiB
    float*  gg    = (float*) (ws + 144 * MB);        // 16 MiB
    __bf16* hh    = (__bf16*)(ws + 160 * MB);        // 32 MiB
    __bf16* xb    = (__bf16*)(ws + 192 * MB);        //  8 MiB
    __bf16* ob    = (__bf16*)(ws + 200 * MB);        //  8 MiB  -> 208 MiB total

    hipMemcpyAsync(x, src, sizeof(float) * (size_t)MROWS * DMODEL,
                   hipMemcpyDeviceToDevice, stream);

    const dim3 blk(256);
    const dim3 cvtg(2048);
    cvt_f32_bf16<<<cvtg, blk, 0, stream>>>(Wqkv, wbqkv, (long)LAYERS * 3 * DMODEL * DMODEL);
    cvt_f32_bf16<<<cvtg, blk, 0, stream>>>(Wo,   wbo,   (long)LAYERS * DMODEL * DMODEL);
    cvt_f32_bf16<<<cvtg, blk, 0, stream>>>(W1,   wb1,   (long)LAYERS * FFNDIM * DMODEL);
    cvt_f32_bf16<<<cvtg, blk, 0, stream>>>(W2,   wb2,   (long)LAYERS * DMODEL * FFNDIM);
    cvt_f32_bf16<<<cvtg, blk, 0, stream>>>(src,  xb,    (long)MROWS * DMODEL);

    for (int l = 0; l < LAYERS; ++l) {
        // QKV projection: [4096,1024] x [3072,1024]^T -> f32
        gemm_wmma<float, false><<<dim3((3 * DMODEL) / BN, MROWS / BM), blk, 0, stream>>>(
            xb, wbqkv + (size_t)l * 3 * DMODEL * DMODEL, bqkv + (size_t)l * 3 * DMODEL,
            qkv, MROWS, 3 * DMODEL, DMODEL);

        // banded softmax attention (8 keys per query), wave per query -> bf16
        attn_band_kernel<<<dim3((BATCH * NHEADS * SEQ) * 32 / 256), blk, 0, stream>>>(qkv, ob);

        // output projection -> f32
        gemm_wmma<float, false><<<dim3(DMODEL / BN, MROWS / BM), blk, 0, stream>>>(
            ob, wbo + (size_t)l * DMODEL * DMODEL, bo + (size_t)l * DMODEL,
            gg, MROWS, DMODEL, DMODEL);

        ln_residual_kernel<<<dim3(MROWS), blk, 0, stream>>>(
            x, gg, ln1w + (size_t)l * DMODEL, ln1b + (size_t)l * DMODEL, x, xb);

        // FFN up + exact GELU -> bf16 hidden
        gemm_wmma<__bf16, true><<<dim3(FFNDIM / BN, MROWS / BM), blk, 0, stream>>>(
            xb, wb1 + (size_t)l * FFNDIM * DMODEL, b1 + (size_t)l * FFNDIM,
            hh, MROWS, FFNDIM, DMODEL);

        // FFN down -> f32
        gemm_wmma<float, false><<<dim3(DMODEL / BN, MROWS / BM), blk, 0, stream>>>(
            hh, wb2 + (size_t)l * DMODEL * FFNDIM, b2 + (size_t)l * DMODEL,
            gg, MROWS, DMODEL, FFNDIM);

        ln_residual_kernel<<<dim3(MROWS), blk, 0, stream>>>(
            x, gg, ln2w + (size_t)l * DMODEL, ln2b + (size_t)l * DMODEL, x, xb);
    }
}
